// MB_Mamba_Block_49426483642710
// MI455X (gfx1250) — matrix operand order repaired and checked by reading, not compiled, not run on hardware
//
#include <hip/hip_runtime.h>
#include <hip/hip_bf16.h>
#include <math.h>

// ---------------------------------------------------------------------------
// Problem constants (fixed by the reference)
// ---------------------------------------------------------------------------
#define BATCH   2
#define LSEQ    2048
#define DMODEL  512
#define DINNER  1024      // EXPAND * D_MODEL
#define DSTATE  16
#define DTRANK  32
#define DCONV   4
#define NROWS   (BATCH * LSEQ)          // 4096 token rows
#define DBCW    (DTRANK + 2 * DSTATE)   // 64

typedef _Float16 v16h __attribute__((ext_vector_type(16)));
typedef _Float16 v8h  __attribute__((ext_vector_type(8)));
typedef float    v8f  __attribute__((ext_vector_type(8)));

enum { EPI_STORE = 0, EPI_BIAS = 1, EPI_BIAS_SOFTPLUS = 2, EPI_ACC = 3 };

// ---------------------------------------------------------------------------
// f32 -> f16 conversion (weights / activations feeding WMMA)
// ---------------------------------------------------------------------------
__global__ void cvt_f32_f16_kernel(const float* __restrict__ src,
                                   _Float16* __restrict__ dst, int n) {
    int i = blockIdx.x * blockDim.x + threadIdx.x;
    if (i < n) dst[i] = (_Float16)src[i];
}

// ---------------------------------------------------------------------------
// LayerNorm over D_MODEL=512; emits f16 (for W_in GEMM) and relu-f16 (gate GEMM)
// ---------------------------------------------------------------------------
__global__ __launch_bounds__(256)
void ln_kernel(const float* __restrict__ x, const float* __restrict__ g,
               const float* __restrict__ b,
               _Float16* __restrict__ fnh, _Float16* __restrict__ fnrh) {
    __shared__ float ssum[256];
    __shared__ float ssq[256];
    const int row = blockIdx.x;
    const int t   = threadIdx.x;
    const float v0 = x[(size_t)row * DMODEL + t];
    const float v1 = x[(size_t)row * DMODEL + t + 256];
    ssum[t] = v0 + v1;
    ssq[t]  = v0 * v0 + v1 * v1;
    __syncthreads();
    for (int s = 128; s > 0; s >>= 1) {
        if (t < s) { ssum[t] += ssum[t + s]; ssq[t] += ssq[t + s]; }
        __syncthreads();
    }
    const float mean = ssum[0] * (1.0f / DMODEL);
    const float var  = ssq[0] * (1.0f / DMODEL) - mean * mean;
    const float rstd = rsqrtf(var + 1e-5f);
    const float y0 = (v0 - mean) * rstd * g[t] + b[t];
    const float y1 = (v1 - mean) * rstd * g[t + 256] + b[t + 256];
    fnh [(size_t)row * DMODEL + t]       = (_Float16)y0;
    fnh [(size_t)row * DMODEL + t + 256] = (_Float16)y1;
    fnrh[(size_t)row * DMODEL + t]       = (_Float16)fmaxf(y0, 0.0f);
    fnrh[(size_t)row * DMODEL + t + 256] = (_Float16)fmaxf(y1, 0.0f);
}

// ---------------------------------------------------------------------------
// WMMA GEMM:  C[M,N] = A[M,K] (f16 row-major, ld=lda) x W[N,K]^T (f16 row-major)
//
// Block = 4 waves; block tile = 64(M) x 64(N).  The B (weight) tile is shared
// by all 4 waves: staged once per workgroup into LDS via the gfx1250 async
// path (global_load_async_to_lds_b128 / ASYNCcnt, double-buffered 2 x 4KB),
// then each lane reads its WMMA B-fragment as one 32B ds_load.  A is
// wave-private -> direct global loads + global_prefetch.  The last K-chunk is
// peeled so the steady-state loop is branch-free.
// flipA / flipC remap l -> L-1-l inside each batch for the backward direction.
// ---------------------------------------------------------------------------
template <int EPI>
__global__ __launch_bounds__(128)
void gemm_wmma_kernel(const _Float16* __restrict__ A,
                      const _Float16* __restrict__ W,
                      const float* __restrict__ bias,
                      float* __restrict__ C,
                      _Float16* __restrict__ Ch,
                      int M, int N, int K, int lda, int flipA, int flipC) {
    __shared__ alignas(64) _Float16 sB[2][64 * 32];   // [buf][row*32 + k]

    const int tid     = threadIdx.x;
    const int lane    = tid & 31;
    const int wave    = tid >> 5;
    const int half_id = lane >> 4;       // 0: lanes 0-15, 1: lanes 16-31
    const int lr      = lane & 15;
    const int tile_m  = blockIdx.x * 4 + wave;
    const int n0      = blockIdx.y * 64;

    // ---- A operand (contiguous K order matching B: lane<16 -> K {0..15},
    // lane>=16 -> K {16..31}); both lane groups cover rows M=0..15.
    int row = tile_m * 16 + lr;
    if (flipA) {
        int bb = row >> 11;              // row / LSEQ
        int ll = row & (LSEQ - 1);
        row = (bb << 11) + (LSEQ - 1 - ll);
    }
    const _Float16* Arow = A + (size_t)row * lda + half_id * 16;

    // ---- async B producer: 128 threads x 2 b128 loads cover 64 rows x 64B
    const int lid0 = tid * 2, lid1 = tid * 2 + 1;
    const int pr0 = lid0 >> 2, ps0 = lid0 & 3;       // row-in-tile, 16B segment
    const int pr1 = lid1 >> 2, ps1 = lid1 & 3;
    const _Float16* gB0 = W + (size_t)(n0 + pr0) * K + ps0 * 8;
    const _Float16* gB1 = W + (size_t)(n0 + pr1) * K + ps1 * 8;
    uint32_t lB0[2], lB1[2];             // LDS byte offsets (low 32b of flat addr)
#pragma unroll
    for (int p = 0; p < 2; ++p) {
        lB0[p] = (uint32_t)(uintptr_t)&sB[p][pr0 * 32 + ps0 * 8];
        lB1[p] = (uint32_t)(uintptr_t)&sB[p][pr1 * 32 + ps1 * 8];
    }
    auto issue = [&](int p, int k0) {
        asm volatile("global_load_async_to_lds_b128 %0, %1, off"
                     :: "v"(lB0[p]), "v"(gB0 + k0) : "memory");
        asm volatile("global_load_async_to_lds_b128 %0, %1, off"
                     :: "v"(lB1[p]), "v"(gB1 + k0) : "memory");
    };

    v8f c0 = {}, c1 = {}, c2 = {}, c3 = {};
    const int kW = half_id * 16;         // B fragment: halfwords kW..kW+15

    auto compute = [&](int buf, int k0, bool prefetch) {
        v8h alo = *(const v8h*)(Arow + k0);
        v8h ahi = *(const v8h*)(Arow + k0 + 8);
        if (prefetch) __builtin_prefetch(Arow + k0 + 32, 0, 1);
        v16h a;
#pragma unroll
        for (int q = 0; q < 8; ++q) { a[q] = alo[q]; a[q + 8] = ahi[q]; }

        v16h b0 = *(const v16h*)&sB[buf][(0 * 16 + lr) * 32 + kW];
        v16h b1 = *(const v16h*)&sB[buf][(1 * 16 + lr) * 32 + kW];
        v16h b2 = *(const v16h*)&sB[buf][(2 * 16 + lr) * 32 + kW];
        v16h b3 = *(const v16h*)&sB[buf][(3 * 16 + lr) * 32 + kW];

        c0 = __builtin_amdgcn_wmma_f32_16x16x32_f16(false, a, false, b0, (short)0, c0, false, false);
        c1 = __builtin_amdgcn_wmma_f32_16x16x32_f16(false, a, false, b1, (short)0, c1, false, false);
        c2 = __builtin_amdgcn_wmma_f32_16x16x32_f16(false, a, false, b2, (short)0, c2, false, false);
        c3 = __builtin_amdgcn_wmma_f32_16x16x32_f16(false, a, false, b3, (short)0, c3, false, false);
    };

    const int nk = K / 32;
    issue(0, 0);                         // prologue: chunk 0 -> buffer 0

    // Steady state: branch-free double-buffered pipeline (last chunk peeled).
    for (int i = 0; i + 1 < nk; ++i) {
        const int k0 = i * 32;
        issue((i + 1) & 1, k0 + 32);     // next chunk -> other buffer
        asm volatile("s_wait_asynccnt 0x2" ::: "memory");   // chunk i landed
        __syncthreads();                 // LDS writes visible to all waves
        compute(i & 1, k0, true);
        __syncthreads();                 // all waves done before buffer reuse
    }
    // Tail: last chunk (covers nk==1, e.g. the K=32 dt GEMM).
    asm volatile("s_wait_asynccnt 0x0" ::: "memory");
    __syncthreads();
    compute((nk - 1) & 1, (nk - 1) * 32, false);

    // Epilogue. C/D layout: VGPR r -> M = r + half_id*8, N = lane&15.
#pragma unroll
    for (int j = 0; j < 4; ++j) {
        v8f& c = (j == 0 ? c0 : j == 1 ? c1 : j == 2 ? c2 : c3);
        const int n = n0 + j * 16 + lr;
        const float bi = (EPI == EPI_BIAS || EPI == EPI_BIAS_SOFTPLUS) ? bias[n] : 0.0f;
#pragma unroll
        for (int r = 0; r < 8; ++r) {
            int m = tile_m * 16 + half_id * 8 + r;
            if (flipC) {
                int bb = m >> 11;
                int ll = m & (LSEQ - 1);
                m = (bb << 11) + (LSEQ - 1 - ll);
            }
            float v = c[r] + bi;
            if (EPI == EPI_BIAS_SOFTPLUS)
                v = (v > 20.0f) ? v : log1pf(expf(v));   // softplus
            const size_t o = (size_t)m * N + n;
            if (EPI == EPI_ACC) C[o] += v; else C[o] = v;
            if (Ch) Ch[o] = (_Float16)v;
        }
    }
}

// ---------------------------------------------------------------------------
// Depthwise causal conv (D_CONV=4) + bias + SiLU over the x half of xz.
// ---------------------------------------------------------------------------
__global__ __launch_bounds__(256)
void conv_silu_kernel(const float* __restrict__ xz,
                      const float* __restrict__ cw,
                      const float* __restrict__ cb,
                      float* __restrict__ xc, _Float16* __restrict__ xch) {
    const int idx = blockIdx.x * 256 + threadIdx.x;     // over NROWS * DINNER
    const int d   = idx & (DINNER - 1);
    const int rw  = idx >> 10;
    const int l   = rw & (LSEQ - 1);
    const int b   = rw >> 11;
    float acc = cb[d];
#pragma unroll
    for (int j = 0; j < DCONV; ++j) {
        const int ls = l - (DCONV - 1) + j;
        if (ls >= 0)
            acc += cw[d * DCONV + j] *
                   xz[((size_t)(b * LSEQ + ls)) * (2 * DINNER) + d];
    }
    const float s = acc / (1.0f + expf(-acc));          // silu
    xc[idx]  = s;
    xch[idx] = (_Float16)s;
}

// ---------------------------------------------------------------------------
// Selective scan. One thread per (batch, channel); 16 recurrent states in
// registers; B_t / C_t broadcast via LDS each timestep. Fuses +D*x and
// *silu(z) and writes the f16 operand for the W_out WMMA GEMM.
// ---------------------------------------------------------------------------
__global__ __launch_bounds__(256)
void scan_kernel(const float* __restrict__ dbc,
                 const float* __restrict__ dtv,
                 const float* __restrict__ xc,
                 const float* __restrict__ xz,
                 const float* __restrict__ A_log,
                 const float* __restrict__ Dp,
                 _Float16* __restrict__ ygh) {
    __shared__ float sBC[2 * DSTATE];
    const int b = blockIdx.x >> 2;
    const int d = ((blockIdx.x & 3) << 8) + threadIdx.x;

    float Areg[DSTATE], h[DSTATE];
#pragma unroll
    for (int n = 0; n < DSTATE; ++n) {
        Areg[n] = -expf(A_log[d * DSTATE + n]);
        h[n] = 0.0f;
    }
    const float Dd = Dp[d];

    for (int l = 0; l < LSEQ; ++l) {
        const size_t r = (size_t)b * LSEQ + l;
        __syncthreads();
        if (threadIdx.x < 2 * DSTATE)
            sBC[threadIdx.x] = dbc[r * DBCW + DTRANK + threadIdx.x];
        __syncthreads();

        const float dt = dtv[r * DINNER + d];
        const float xv = xc[r * DINNER + d];
        const float dx = dt * xv;
        float y = 0.0f;
#pragma unroll
        for (int n = 0; n < DSTATE; ++n) {
            h[n] = expf(dt * Areg[n]) * h[n] + dx * sBC[n];
            y += h[n] * sBC[DSTATE + n];
        }
        const float z  = xz[r * (2 * DINNER) + DINNER + d];
        const float yo = (y + Dd * xv) * (z / (1.0f + expf(-z)));
        ygh[r * DINNER + d] = (_Float16)yo;
    }
}

// ---------------------------------------------------------------------------
// Final: out = fw * (out_f + out_b)
// ---------------------------------------------------------------------------
__global__ void final_mul_kernel(const float* __restrict__ fw,
                                 const float* __restrict__ acc,
                                 float* __restrict__ out, int n) {
    int i = blockIdx.x * blockDim.x + threadIdx.x;
    if (i < n) out[i] = fw[i] * acc[i];
}

// ---------------------------------------------------------------------------
// Host launcher
// ---------------------------------------------------------------------------
extern "C" void kernel_launch(void* const* d_in, const int* in_sizes, int n_in,
                              void* d_out, int out_size, void* d_ws, size_t ws_size,
                              hipStream_t stream) {
    (void)in_sizes; (void)n_in; (void)out_size; (void)ws_size;

    const float* x      = (const float*)d_in[0];
    const float* ln_g   = (const float*)d_in[1];
    const float* ln_b   = (const float*)d_in[2];
    const float* gate_w = (const float*)d_in[3];
    const float* gate_b = (const float*)d_in[4];

    // Workspace bump allocator (deterministic layout, ~110 MB)
    char* wp = (char*)d_ws;
    auto alloc = [&](size_t bytes) -> char* {
        char* p = wp;
        wp += (bytes + 255) & ~(size_t)255;
        return p;
    };
    _Float16* fnh    = (_Float16*)alloc((size_t)NROWS * DMODEL * 2);
    _Float16* fnrh   = (_Float16*)alloc((size_t)NROWS * DMODEL * 2);
    float*    fw     = (float*)   alloc((size_t)NROWS * DMODEL * 4);
    float*    outacc = (float*)   alloc((size_t)NROWS * DMODEL * 4);
    _Float16* gwh    = (_Float16*)alloc((size_t)DMODEL * DMODEL * 2);
    _Float16* winh   = (_Float16*)alloc((size_t)2 * DINNER * DMODEL * 2);
    _Float16* wxh    = (_Float16*)alloc((size_t)DBCW * DINNER * 2);
    _Float16* wdth   = (_Float16*)alloc((size_t)DINNER * DTRANK * 2);
    _Float16* wouth  = (_Float16*)alloc((size_t)DMODEL * DINNER * 2);
    float*    xz     = (float*)   alloc((size_t)NROWS * 2 * DINNER * 4);
    float*    xc     = (float*)   alloc((size_t)NROWS * DINNER * 4);
    _Float16* xch    = (_Float16*)alloc((size_t)NROWS * DINNER * 2);
    float*    dbc    = (float*)   alloc((size_t)NROWS * DBCW * 4);
    _Float16* dbch   = (_Float16*)alloc((size_t)NROWS * DBCW * 2);
    float*    dtb    = (float*)   alloc((size_t)NROWS * DINNER * 4);
    _Float16* ygh    = (_Float16*)alloc((size_t)NROWS * DINNER * 2);

    auto cvt = [&](const float* s, _Float16* d, int n) {
        cvt_f32_f16_kernel<<<(n + 255) / 256, 256, 0, stream>>>(s, d, n);
    };

    // LayerNorm + gate path
    ln_kernel<<<NROWS, 256, 0, stream>>>(x, ln_g, ln_b, fnh, fnrh);
    cvt(gate_w, gwh, DMODEL * DMODEL);
    gemm_wmma_kernel<EPI_BIAS><<<dim3(NROWS / 64, DMODEL / 64), 128, 0, stream>>>(
        fnrh, gwh, gate_b, fw, nullptr, NROWS, DMODEL, DMODEL, DMODEL, 0, 0);

    // Two Mamba directions (dir=1 runs on the time-flipped sequence)
    for (int dir = 0; dir < 2; ++dir) {
        const float* W_in   = (const float*)d_in[5 + 9 * dir + 0];
        const float* conv_w = (const float*)d_in[5 + 9 * dir + 1];
        const float* conv_b = (const float*)d_in[5 + 9 * dir + 2];
        const float* W_x    = (const float*)d_in[5 + 9 * dir + 3];
        const float* W_dt   = (const float*)d_in[5 + 9 * dir + 4];
        const float* b_dt   = (const float*)d_in[5 + 9 * dir + 5];
        const float* A_log  = (const float*)d_in[5 + 9 * dir + 6];
        const float* Dp     = (const float*)d_in[5 + 9 * dir + 7];
        const float* W_out  = (const float*)d_in[5 + 9 * dir + 8];

        cvt(W_in,  winh,  2 * DINNER * DMODEL);
        cvt(W_x,   wxh,   DBCW * DINNER);
        cvt(W_dt,  wdth,  DINNER * DTRANK);
        cvt(W_out, wouth, DMODEL * DINNER);

        // xz = u @ W_in^T   (flipped input rows for backward dir)
        gemm_wmma_kernel<EPI_STORE><<<dim3(NROWS / 64, (2 * DINNER) / 64), 128, 0, stream>>>(
            fnh, winh, nullptr, xz, nullptr, NROWS, 2 * DINNER, DMODEL, DMODEL, dir, 0);

        // depthwise causal conv + silu
        conv_silu_kernel<<<(NROWS * DINNER) / 256, 256, 0, stream>>>(
            xz, conv_w, conv_b, xc, xch);

        // dbc = xc @ W_x^T (also emit f16 copy for the dt GEMM)
        gemm_wmma_kernel<EPI_STORE><<<dim3(NROWS / 64, DBCW / 64), 128, 0, stream>>>(
            xch, wxh, nullptr, dbc, dbch, NROWS, DBCW, DINNER, DINNER, 0, 0);

        // dt = softplus(dbc[:, :32] @ W_dt^T + b_dt)
        gemm_wmma_kernel<EPI_BIAS_SOFTPLUS><<<dim3(NROWS / 64, DINNER / 64), 128, 0, stream>>>(
            dbch, wdth, b_dt, dtb, nullptr, NROWS, DINNER, DTRANK, DBCW, 0, 0);

        // selective scan (fused +D*x and *silu(z))
        scan_kernel<<<(BATCH * DINNER) / 256, 256, 0, stream>>>(
            dbc, dtb, xc, xz, A_log, Dp, ygh);

        // out_dir = yg @ W_out^T  (backward: un-flip rows and accumulate)
        if (dir == 0)
            gemm_wmma_kernel<EPI_STORE><<<dim3(NROWS / 64, DMODEL / 64), 128, 0, stream>>>(
                ygh, wouth, nullptr, outacc, nullptr, NROWS, DMODEL, DINNER, DINNER, 0, 0);
        else
            gemm_wmma_kernel<EPI_ACC><<<dim3(NROWS / 64, DMODEL / 64), 128, 0, stream>>>(
                ygh, wouth, nullptr, outacc, nullptr, NROWS, DMODEL, DINNER, DINNER, 0, 1);
    }

    // out = fw * (out_f + out_b)
    final_mul_kernel<<<(NROWS * DMODEL + 255) / 256, 256, 0, stream>>>(
        fw, outacc, (float*)d_out, NROWS * DMODEL);
}
